// GraphCNN_69191923138906
// MI455X (gfx1250) — matrix operand order repaired and checked by reading, not compiled, not run on hardware
//
#include <hip/hip_runtime.h>
#include <hip/hip_bf16.h>

typedef __attribute__((ext_vector_type(16))) _Float16 v16h;
typedef __attribute__((ext_vector_type(8)))  float    v8f;

#define D128   128
#define NGRAPH 128
#define OUTD   64
#define NLAYER 4
#define BN_EPS 1e-5f
#define SA     132   // padded LDS stride in halves (132*2B = 264B -> bank rotate)

union Frag { v16h v; _Float16 e[16]; };

// ---------------------------------------------------------------- zero
__global__ __launch_bounds__(256) void zero_f32(float* p, int n) {
    int i = blockIdx.x * 256 + threadIdx.x;
    if (i < n) p[i] = 0.0f;
}

// ------------------------------------------------- weight f16 transpose
// Wt layout per matrix: [n*128 + k] = W[k*128 + n]  (B is K x N in the GEMM)
__global__ __launch_bounds__(256) void prep_weights(
    const float* __restrict__ W1, const float* __restrict__ W2,
    const float* __restrict__ Pw, _Float16* __restrict__ Wt,
    _Float16* __restrict__ Pt)
{
    int i = blockIdx.x * 256 + threadIdx.x;
    const int WN = NLAYER * D128 * D128;
    if (i < WN) {
        int l = i / (D128 * D128), r = i % (D128 * D128);
        int k = r / D128, n = r % D128;
        Wt[(2 * l) * (D128 * D128) + n * D128 + k]     = (_Float16)W1[i];
        Wt[(2 * l + 1) * (D128 * D128) + n * D128 + k] = (_Float16)W2[i];
    } else {
        int j = i - WN;
        if (j < (NLAYER + 1) * D128 * OUTD) {
            int l = j / (D128 * OUTD), r = j % (D128 * OUTD);
            int k = r / OUTD, n = r % OUTD;
            Pt[l * (D128 * OUTD) + n * D128 + k] = (_Float16)Pw[j];
        }
    }
}

// ---------------------------------------------------------- edge scatter
// one wave32 per edge; lane handles 4 features
__global__ __launch_bounds__(256) void edge_agg(
    const float* __restrict__ h, const int* __restrict__ src,
    const int* __restrict__ dst, float* agg, int nE)
{
    int w = threadIdx.x >> 5;
    int lane = threadIdx.x & 31;
    int e = blockIdx.x * 8 + w;
    if (e >= nE) return;
    int s = src[e];
    int d = dst[e];
    float4 v = *((const float4*)(h + (size_t)s * D128) + lane);
    float* ap = agg + (size_t)d * D128 + lane * 4;
    atomicAdd(ap + 0, v.x);
    atomicAdd(ap + 1, v.y);
    atomicAdd(ap + 2, v.z);
    atomicAdd(ap + 3, v.w);
}

// ------------------------------------------- fused GEMM + BN statistics
// block: 128 rows x 128 cols, 8 waves; wave w -> rows [16w,16w+16), 8 col tiles
// pooled = agg + (1+eps)*h  (useAgg) or just hin;  z = pooled @ W + bias
// accumulates column sum / sum-of-squares for BatchNorm into gsum/gsq
__global__ __launch_bounds__(256) void gemm_bn_stats(
    const float* hin, const float* agg, const _Float16* __restrict__ Wt,
    const float* __restrict__ bias, const float* __restrict__ epsArr,
    int layer, int useAgg, float* zout, float* gsum, float* gsq, int N)
{
    __shared__ _Float16 ldsA[D128 * SA];
    __shared__ _Float16 ldsW[D128 * SA];
    __shared__ float ssum[D128];
    __shared__ float ssq[D128];

    int t = threadIdx.x;
    int row0 = blockIdx.x * D128;
    float cf = useAgg ? (1.0f + epsArr[layer]) : 1.0f;

    // stage activation tile (f32 -> f16), zero-pad rows >= N
    #pragma unroll
    for (int i = 0; i < 16; ++i) {
        int idx = t + 256 * i;          // 4096 float4 slots
        int r   = idx >> 5;             // 0..127
        int c4  = (idx & 31) * 4;
        int rg  = row0 + r;
        float4 v = {0.f, 0.f, 0.f, 0.f};
        if (rg < N) {
            v = *(const float4*)(hin + (size_t)rg * D128 + c4);
            if (useAgg) {
                float4 a = *(const float4*)(agg + (size_t)rg * D128 + c4);
                v.x = a.x + cf * v.x; v.y = a.y + cf * v.y;
                v.z = a.z + cf * v.z; v.w = a.w + cf * v.w;
            }
        }
        _Float16* d = &ldsA[r * SA + c4];
        d[0] = (_Float16)v.x; d[1] = (_Float16)v.y;
        d[2] = (_Float16)v.z; d[3] = (_Float16)v.w;
    }
    // stage transposed weights (already f16): rows n, cols k
    #pragma unroll
    for (int i = 0; i < 8; ++i) {
        int idx = t + 256 * i;          // 2048 8-half chunks
        int r   = idx >> 4;
        int c8  = (idx & 15) * 8;
        uint4 wv = *(const uint4*)(Wt + r * D128 + c8);
        *(uint2*)&ldsW[r * SA + c8]     = make_uint2(wv.x, wv.y);
        *(uint2*)&ldsW[r * SA + c8 + 4] = make_uint2(wv.z, wv.w);
    }
    if (t < D128) { ssum[t] = 0.f; ssq[t] = 0.f; }
    __syncthreads();

    int lane = t & 31, w = t >> 5;
    int m = lane & 15, hi = lane >> 4;
    int arow = 16 * w + m;

    v8f acc[8] = {};
    for (int kb = 0; kb < D128; kb += 32) {
        Frag a;
        #pragma unroll
        for (int vv = 0; vv < 8; ++vv) {
            int k = kb + hi * 8 + (vv & 3) * 2 + (vv >> 2) * 16;
            a.e[2 * vv]     = ldsA[arow * SA + k];
            a.e[2 * vv + 1] = ldsA[arow * SA + k + 1];
        }
        #pragma unroll
        for (int tt = 0; tt < 8; ++tt) {
            Frag b;
            int n   = tt * 16 + m;
            int kb2 = kb + hi * 8;
            #pragma unroll
            for (int e = 0; e < 16; ++e) b.e[e] = ldsW[n * SA + kb2 + e + (e >> 3) * 8];
            acc[tt] = __builtin_amdgcn_wmma_f32_16x16x32_f16(
                false, a.v, false, b.v, (short)0, acc[tt], false, false);
        }
    }

    // epilogue: bias, store z, reduce col sums for BN
    #pragma unroll
    for (int tt = 0; tt < 8; ++tt) {
        int col = tt * 16 + m;
        float bv = bias[col];
        float cs = 0.f, cq = 0.f;
        #pragma unroll
        for (int r = 0; r < 8; ++r) {
            int rg = row0 + 16 * w + hi * 8 + r;
            float val = acc[tt][r] + bv;
            if (rg < N) {
                zout[(size_t)rg * D128 + col] = val;
                cs += val; cq += val * val;
            }
        }
        cs += __shfl_xor(cs, 16, 32);
        cq += __shfl_xor(cq, 16, 32);
        if (hi == 0) { atomicAdd(&ssum[col], cs); atomicAdd(&ssq[col], cq); }
    }
    __syncthreads();
    if (t < D128) { atomicAdd(&gsum[t], ssum[t]); atomicAdd(&gsq[t], ssq[t]); }
}

// ------------------------------------------------------------- BN + ReLU
__global__ __launch_bounds__(256) void bn_relu(
    const float* __restrict__ z, float* __restrict__ out,
    const float* __restrict__ gsum, const float* __restrict__ gsq,
    const float* __restrict__ gamma, const float* __restrict__ beta,
    float invN, int nElem)
{
    int i = blockIdx.x * 256 + threadIdx.x;
    if (i >= nElem) return;
    int col = i & (D128 - 1);
    float mu  = gsum[col] * invN;
    float var = gsq[col] * invN - mu * mu;
    float sc  = gamma[col] * rsqrtf(var + BN_EPS);
    float v   = (z[i] - mu) * sc + beta[col];
    out[i] = v > 0.f ? v : 0.f;
}

// ------------------------------------------- per-graph sum pooling (sorted ids)
__global__ __launch_bounds__(256) void graph_pool(
    const float* __restrict__ h, const int* __restrict__ gid,
    float* pg, int N)
{
    const int NB = 512;
    __shared__ float sacc[16 * D128];
    int n0   = blockIdx.x * NB;
    int nEnd = min(n0 + NB, N);
    int t = threadIdx.x;
    int g0 = gid[n0];
    int g1 = gid[nEnd - 1];
    int ng = g1 - g0 + 1;

    if (ng <= 16) {
        for (int i = t; i < 16 * D128; i += 256) sacc[i] = 0.f;
        __syncthreads();
        for (int i = 0; i < NB * D128 / 256; ++i) {
            int flat = i * 256 + t;
            int node = n0 + (flat >> 7);
            if (node < nEnd) {
                int col = flat & 127;
                int g = gid[node];
                atomicAdd(&sacc[(g - g0) * D128 + col],
                          h[(size_t)node * D128 + col]);
            }
        }
        __syncthreads();
        for (int i = t; i < ng * D128; i += 256)
            atomicAdd(&pg[g0 * D128 + i], sacc[i]);
    } else {
        for (int i = 0; i < NB * D128 / 256; ++i) {
            int flat = i * 256 + t;
            int node = n0 + (flat >> 7);
            if (node < nEnd) {
                int col = flat & 127;
                atomicAdd(&pg[gid[node] * D128 + col],
                          h[(size_t)node * D128 + col]);
            }
        }
    }
}

// -------------------------------- readout: score = sum_l pg[l] @ predW[l] + b
// single block, 8 waves; K accumulated over all 5 layers (K = 640)
__global__ __launch_bounds__(256) void readout(
    const float* __restrict__ pg, const _Float16* __restrict__ Pt,
    const float* __restrict__ predb, float* __restrict__ out)
{
    __shared__ _Float16 ldsA[D128 * SA];
    __shared__ _Float16 ldsB[OUTD * SA];
    int t = threadIdx.x;
    int lane = t & 31, w = t >> 5;
    int m = lane & 15, hi = lane >> 4;
    int arow = 16 * w + m;

    v8f acc[4] = {};
    for (int l = 0; l < NLAYER + 1; ++l) {
        __syncthreads();
        // stage pooled-graph matrix (128 x 128 f32 -> f16)
        #pragma unroll
        for (int i = 0; i < 16; ++i) {
            int idx = t + 256 * i;
            int r = idx >> 5, c4 = (idx & 31) * 4;
            float4 v = *(const float4*)(pg + l * D128 * D128 + r * D128 + c4);
            _Float16* d = &ldsA[r * SA + c4];
            d[0] = (_Float16)v.x; d[1] = (_Float16)v.y;
            d[2] = (_Float16)v.z; d[3] = (_Float16)v.w;
        }
        // stage predW^T chunk (64 rows x 128 k, f16)
        #pragma unroll
        for (int i = 0; i < 4; ++i) {
            int idx = t + 256 * i;      // 1024 8-half chunks
            int r = idx >> 4, c8 = (idx & 15) * 8;
            uint4 wv = *(const uint4*)(Pt + l * D128 * OUTD + r * D128 + c8);
            *(uint2*)&ldsB[r * SA + c8]     = make_uint2(wv.x, wv.y);
            *(uint2*)&ldsB[r * SA + c8 + 4] = make_uint2(wv.z, wv.w);
        }
        __syncthreads();
        for (int kb = 0; kb < D128; kb += 32) {
            Frag a;
            #pragma unroll
            for (int vv = 0; vv < 8; ++vv) {
                int k = kb + hi * 8 + (vv & 3) * 2 + (vv >> 2) * 16;
                a.e[2 * vv]     = ldsA[arow * SA + k];
                a.e[2 * vv + 1] = ldsA[arow * SA + k + 1];
            }
            #pragma unroll
            for (int tt = 0; tt < 4; ++tt) {
                Frag b;
                int n = tt * 16 + m;
                int kb2 = kb + hi * 8;
                #pragma unroll
                for (int e = 0; e < 16; ++e) b.e[e] = ldsB[n * SA + kb2 + e + (e >> 3) * 8];
                acc[tt] = __builtin_amdgcn_wmma_f32_16x16x32_f16(
                    false, a.v, false, b.v, (short)0, acc[tt], false, false);
            }
        }
    }
    #pragma unroll
    for (int tt = 0; tt < 4; ++tt) {
        int col = tt * 16 + m;
        float bsum = 0.f;
        #pragma unroll
        for (int l = 0; l < NLAYER + 1; ++l) bsum += predb[l * OUTD + col];
        #pragma unroll
        for (int r = 0; r < 8; ++r) {
            int row = 16 * w + hi * 8 + r;
            out[row * OUTD + col] = acc[tt][r] + bsum;
        }
    }
}

// ----------------------------------------------------------------- driver
extern "C" void kernel_launch(void* const* d_in, const int* in_sizes, int n_in,
                              void* d_out, int out_size, void* d_ws, size_t ws_size,
                              hipStream_t stream) {
    const float* x     = (const float*)d_in[0];
    const int*   esrc  = (const int*)d_in[1];
    const int*   edst  = (const int*)d_in[2];
    const int*   gids  = (const int*)d_in[3];
    const float* eps   = (const float*)d_in[4];
    const float* W1    = (const float*)d_in[5];
    const float* b1    = (const float*)d_in[6];
    const float* g1    = (const float*)d_in[7];
    const float* be1   = (const float*)d_in[8];
    const float* W2    = (const float*)d_in[9];
    const float* b2    = (const float*)d_in[10];
    const float* g2    = (const float*)d_in[11];
    const float* be2   = (const float*)d_in[12];
    const float* predW = (const float*)d_in[13];
    const float* predb = (const float*)d_in[14];

    const int N  = in_sizes[0] / D128;
    const int nE = in_sizes[1];

    // workspace carve (256B aligned)
    char* base = (char*)d_ws;
    size_t off = 0;
    auto carve = [&](size_t bytes) -> void* {
        void* p = base + off;
        off += (bytes + 255) & ~(size_t)255;
        return p;
    };
    float*    B0    = (float*)carve((size_t)N * D128 * 4);           // node features
    float*    B1    = (float*)carve((size_t)N * D128 * 4);           // agg / z / a
    _Float16* Wt    = (_Float16*)carve(2 * NLAYER * D128 * D128 * 2);
    _Float16* Pt    = (_Float16*)carve((NLAYER + 1) * D128 * OUTD * 2);
    float*    stats = (float*)carve(2 * D128 * 4);
    float*    pgbuf = (float*)carve((size_t)(NLAYER + 1) * NGRAPH * D128 * 4);
    float* gsum = stats;
    float* gsq  = stats + D128;

    const int nElem = N * D128;
    const float invN = 1.0f / (float)N;
    dim3 blk(256);

    // weights -> f16 transposed
    {
        int tot = NLAYER * D128 * D128 + (NLAYER + 1) * D128 * OUTD;
        prep_weights<<<dim3((tot + 255) / 256), blk, 0, stream>>>(W1, W2, predW, Wt, Pt);
    }
    // zero pooled-graph accumulators; pool hidden[0] = x
    {
        int tot = (NLAYER + 1) * NGRAPH * D128;
        zero_f32<<<dim3((tot + 255) / 256), blk, 0, stream>>>(pgbuf, tot);
        graph_pool<<<dim3((N + 511) / 512), blk, 0, stream>>>(x, gids, pgbuf, N);
    }

    const float* h = x;
    const int gemmGrid = (N + D128 - 1) / D128;
    for (int l = 0; l < NLAYER; ++l) {
        // aggregate neighbors into B1
        zero_f32<<<dim3((nElem + 255) / 256), blk, 0, stream>>>(B1, nElem);
        edge_agg<<<dim3((nE + 7) / 8), blk, 0, stream>>>(h, esrc, edst, B1, nE);

        // z1 = (agg + (1+eps)h) @ W1 + b1   (in-place into B1) + BN stats
        zero_f32<<<dim3(1), blk, 0, stream>>>(stats, 2 * D128);
        gemm_bn_stats<<<dim3(gemmGrid), blk, 0, stream>>>(
            h, B1, Wt + (2 * l) * D128 * D128, b1 + l * D128, eps, l, 1,
            B1, gsum, gsq, N);
        bn_relu<<<dim3((nElem + 255) / 256), blk, 0, stream>>>(
            B1, B1, gsum, gsq, g1 + l * D128, be1 + l * D128, invN, nElem);

        // z2 = a1 @ W2 + b2 (in-place) + BN stats
        zero_f32<<<dim3(1), blk, 0, stream>>>(stats, 2 * D128);
        gemm_bn_stats<<<dim3(gemmGrid), blk, 0, stream>>>(
            B1, nullptr, Wt + (2 * l + 1) * D128 * D128, b2 + l * D128, eps, l, 0,
            B1, gsum, gsq, N);
        bn_relu<<<dim3((nElem + 255) / 256), blk, 0, stream>>>(
            B1, B0, gsum, gsq, g2 + l * D128, be2 + l * D128, invN, nElem);

        h = B0;
        graph_pool<<<dim3((N + 511) / 512), blk, 0, stream>>>(
            h, gids, pgbuf + (l + 1) * NGRAPH * D128, N);
    }

    readout<<<dim3(1), blk, 0, stream>>>(pgbuf, Pt, predb, (float*)d_out);
}
